// Vega7Model_70300024701623
// MI455X (gfx1250) — matrix operand order repaired and checked by reading, not compiled, not run on hardware
//
#include <hip/hip_runtime.h>
#include <stdint.h>

// Model dims
#define H   512
#define AC  384
#define BB  2
#define TT  32
#define LL  4
#define FFD 2048
#define VV  32000
#define NP  896   // A + H

typedef __attribute__((ext_vector_type(16))) __bf16 v16bf;
typedef __attribute__((ext_vector_type(8)))  __bf16 v8bf;
typedef __attribute__((ext_vector_type(4)))  __bf16 v4bf;
typedef __attribute__((ext_vector_type(8)))  float  v8f;
typedef __attribute__((ext_vector_type(4)))  unsigned int v4u;
typedef __attribute__((ext_vector_type(8)))  int v8i;
typedef __attribute__((ext_vector_type(4)))  int v4i;

// ---------------- workspace layout (bytes) ----------------
#define OFF_STATES 0u                                  // L*B*H*H f32     = 8,388,608
#define OFF_HEADW  (OFF_STATES + 8388608u)             // V*H bf16        = 32,768,000
#define OFF_WG     (OFF_HEADW + 32768000u)             // L*H*H bf16      = 2,097,152
#define OFF_WT     (OFF_WG    + 2097152u)              // L*H*H bf16
#define OFF_X      (OFF_WT    + 2097152u)              // B*H f32
#define OFF_P      (OFF_X     + 4096u)                 // B*NP f32
#define OFF_XLN2   (OFF_P     + 7168u)                 // B*H f32
#define OFF_H1     (OFF_XLN2  + 4096u)                 // B*FF f32
#define OFF_OUTS   (OFF_H1    + 16384u)                // B*T*H f32
#define OFF_XH     (OFF_OUTS  + 131072u)               // 64*H bf16
#define OFF_GTMP   (OFF_XH    + 65536u)                // 256*H f32 ghost tmp

// ---------------- f32 -> bf16 weight conversion ----------------
__global__ void k_cvt_bf16(const float* __restrict__ s, __bf16* __restrict__ d, int n) {
  int i = (blockIdx.x * blockDim.x + threadIdx.x) * 4;
  if (i < n) {
    float4 v = *(const float4*)(s + i);
    v4bf o; o[0] = (__bf16)v.x; o[1] = (__bf16)v.y; o[2] = (__bf16)v.z; o[3] = (__bf16)v.w;
    *(v4bf*)(d + i) = o;
  }
}

// ---------------- per-(t,l) pre: embed(l==0) + LN(x) + p = ln(x)@Wp^T + bp ----------------
__global__ void k_pre(const int* __restrict__ ids, const float* __restrict__ emb,
                      float* __restrict__ x,
                      const float* __restrict__ nxw, const float* __restrict__ nxb,
                      const float* __restrict__ Wp, const float* __restrict__ bp,
                      float* __restrict__ p, int t, int load_emb) {
  __shared__ float xs[BB * H];
  __shared__ float r1[512], r2[512];
  __shared__ float mstat, vstat;
  int tid = threadIdx.x; // blockDim = 896
  if (load_emb) {
    for (int idx = tid; idx < BB * H; idx += blockDim.x) {
      int b = idx >> 9, h = idx & (H - 1);
      x[idx] = emb[(long)ids[b * TT + t] * H + h];
    }
  }
  __syncthreads();
  for (int b = 0; b < BB; ++b) {
    float v = 0.f;
    if (tid < H) { v = x[b * H + tid]; r1[tid] = v; r2[tid] = v * v; }
    __syncthreads();
    for (int off = 256; off > 0; off >>= 1) {
      if (tid < off) { r1[tid] += r1[tid + off]; r2[tid] += r2[tid + off]; }
      __syncthreads();
    }
    if (tid == 0) { float m = r1[0] / H; mstat = m; vstat = r2[0] / H - m * m; }
    __syncthreads();
    if (tid < H)
      xs[b * H + tid] = (v - mstat) * rsqrtf(vstat + 1e-5f) * nxw[tid] + nxb[tid];
    __syncthreads();
  }
  for (int j = tid; j < BB * NP; j += blockDim.x) {
    int b = j / NP, n = j - b * NP;
    const float* w  = Wp + (long)n * H;
    const float* xv = xs + b * H;
    float s = bp[n];
    for (int k = 0; k < H; k += 4)
      s += xv[k] * w[k] + xv[k+1] * w[k+1] + xv[k+2] * w[k+2] + xv[k+3] * w[k+3];
    p[b * NP + n] = s;
  }
}

// ---------------- ghost recurrence (TDM stage + WMMA bf16) + optional rank-1 update ----------
// dst = src + sigmoid(src@Wg^T) * tanh(src@Wt^T) over the 2*128 ghost rows.
// Blocks [0,16): blockIdx = m-tile. TDM DMAs the 16x512 f32 A tile (one contiguous 32KB
// region) into LDS; 8 waves (one 16x64 tile each) run the K loop from LDS.
// Blocks [16,16+1536) when do_outer: state_active += u1 (x) v1 + u2 (x) v2.
__global__ void k_ghost(const float* __restrict__ src0, const float* __restrict__ src1,
                        float* __restrict__ dst0, float* __restrict__ dst1,
                        const __bf16* __restrict__ Wgb, const __bf16* __restrict__ Wtb,
                        float* __restrict__ st_l,
                        const float* __restrict__ p, const float* __restrict__ bp,
                        int do_outer) {
  __shared__ float sA[16 * H];   // 32KB staged A tile (f32)
  int blk = blockIdx.x;
  if (blk >= 16) {
    if (!do_outer) return;
    int idx = (blk - 16) * blockDim.x + threadIdx.x;   // < 2*384*512
    int b = idx / (AC * H);
    int rem = idx - b * (AC * H);
    int i = rem >> 9, j = rem & (H - 1);
    float u1 = p[b * NP + i], v1 = p[b * NP + AC + j];
    float u2 = bp[i],         v2 = bp[AC + j];
    st_l[((long)b * H + i) * H + j] += u1 * v1 + u2 * v2;
    return;
  }
  int lane = threadIdx.x & 31;
  int wv   = threadIdx.x >> 5;
  int mtile = blk;                  // 16 M-tiles
  int ngrp  = wv;                   // 8 N-groups of 64 per block
  int m0 = mtile * 16, n0 = ngrp * 64;
  const float* srcB = (m0 < 128) ? src0 : src1;
  float*       dstB = (m0 < 128) ? dst0 : dst1;
  int lr0  = m0 & 127;
  int half = lane >> 4;

  // ---- Tensor Data Mover: DMA 16 contiguous rows (32KB) global -> LDS (wave 0 only) ----
  if (threadIdx.x < 32) {
    unsigned long long ga = (unsigned long long)(uintptr_t)(srcB + (long)lr0 * H);
    unsigned lds_off = (unsigned)(uintptr_t)sA;
    v4u g0;
    g0[0] = 1u;                                            // count=1, user mode
    g0[1] = lds_off;                                       // lds_addr (bytes)
    g0[2] = (unsigned)ga;                                  // global_addr[31:0]
    g0[3] = (unsigned)((ga >> 32) & 0x01FFFFFFu) | (2u << 30); // addr[56:32] | type=2
    v8i g1;
    g1[0] = (int)(2u << 16);          // workgroup_mask=0, data_size=2 (4 bytes)
    g1[1] = (int)(8192u << 16);       // tensor_dim0[15:0]=8192 -> word1[31:16]
    g1[2] = (int)(1u << 16);          // tensor_dim0 hi=0; tensor_dim1[15:0]=1
    g1[3] = (int)(8192u << 16);       // tensor_dim1 hi=0; tile_dim0=8192 (1-D tile)
    g1[4] = 0;                        // tile_dim1=0 (unused), tile_dim2=0
    g1[5] = 8192;                     // tensor_dim0_stride[31:0]
    g1[6] = 0;                        // stride hi, dim1_stride lo
    g1[7] = 0;
    v4i g2 = {0, 0, 0, 0};
    v4i g3 = {0, 0, 0, 0};
    v8i g4 = {0, 0, 0, 0, 0, 0, 0, 0};
    __builtin_amdgcn_tensor_load_to_lds(g0, g1, g2, g3, g4, 0);
    __builtin_amdgcn_s_wait_tensorcnt(0);
  }
  __syncthreads();

  const float* ar = sA + (lane & 15) * H + half * 16;
  const __bf16* bg[4]; const __bf16* bt[4];
  #pragma unroll
  for (int nt = 0; nt < 4; ++nt) {
    int col = n0 + nt * 16 + (lane & 15);
    bg[nt] = Wgb + (long)col * H + half * 16;
    bt[nt] = Wtb + (long)col * H + half * 16;
  }
  v8f accg[4] = {}; v8f acct[4] = {};
  for (int kk = 0; kk < H; kk += 32) {
    // A fragment from LDS, convert f32 -> bf16 in-register
    v8f lo = *(const v8f*)(ar + kk);
    v8f hi = *(const v8f*)(ar + kk + 8);
    v16bf a;
    #pragma unroll
    for (int i = 0; i < 8; ++i) { a[i] = (__bf16)lo[i]; a[8 + i] = (__bf16)hi[i]; }
    // batch all 8 B fragments first so loads overlap, then issue the 8 WMMAs
    v16bf vbg[4], vbt[4];
    #pragma unroll
    for (int nt = 0; nt < 4; ++nt) vbg[nt] = *(const v16bf*)(bg[nt] + kk);
    #pragma unroll
    for (int nt = 0; nt < 4; ++nt) vbt[nt] = *(const v16bf*)(bt[nt] + kk);
    #pragma unroll
    for (int nt = 0; nt < 4; ++nt)
      accg[nt] = __builtin_amdgcn_wmma_f32_16x16x32_bf16(false, a, false, vbg[nt], (short)0,
                                                         accg[nt], false, false);
    #pragma unroll
    for (int nt = 0; nt < 4; ++nt)
      acct[nt] = __builtin_amdgcn_wmma_f32_16x16x32_bf16(false, a, false, vbt[nt], (short)0,
                                                         acct[nt], false, false);
  }
  int coll = lane & 15;
  #pragma unroll
  for (int nt = 0; nt < 4; ++nt) {
    int col = n0 + nt * 16 + coll;
    #pragma unroll
    for (int r = 0; r < 8; ++r) {
      int rm = r + half * 8;                       // row within the 16-row tile
      float gate = 1.f / (1.f + expf(-accg[nt][r]));
      float tr   = tanhf(acct[nt][r]);
      dstB[(long)(lr0 + rm) * H + col] = sA[rm * H + col] + gate * tr;
    }
  }
}

// ---------------- x += state.mean(rows); xln2 = LN(x) ----------------
__global__ void k_mid(const float* __restrict__ st_l, float* __restrict__ x,
                      const float* __restrict__ nsw, const float* __restrict__ nsb,
                      float* __restrict__ xln2) {
  __shared__ float r1[512], r2[512];
  __shared__ float mstat, vstat;
  int b = blockIdx.x, h = threadIdx.x; // blockDim = 512
  const float* col = st_l + (long)b * H * H + h;
  float s = 0.f;
  for (int r = 0; r < H; ++r) s += col[(long)r * H];
  float xv = x[b * H + h] + s * (1.0f / H);
  x[b * H + h] = xv;
  r1[h] = xv; r2[h] = xv * xv;
  __syncthreads();
  for (int off = 256; off > 0; off >>= 1) {
    if (h < off) { r1[h] += r1[h + off]; r2[h] += r2[h + off]; }
    __syncthreads();
  }
  if (h == 0) { float m = r1[0] / H; mstat = m; vstat = r2[0] / H - m * m; }
  __syncthreads();
  xln2[b * H + h] = (xv - mstat) * rsqrtf(vstat + 1e-5f) * nsw[h] + nsb[h];
}

// ---------------- FFN ----------------
__global__ void k_ffn1(const float* __restrict__ xln2, const float* __restrict__ W1,
                       const float* __restrict__ b1, float* __restrict__ h1) {
  int j = blockIdx.x * blockDim.x + threadIdx.x; // < 4096
  int b = j >> 11, o = j & (FFD - 1);
  const float* w  = W1 + (long)o * H;
  const float* xv = xln2 + b * H;
  float s = b1[o];
  for (int k = 0; k < H; k += 4) {
    float4 wv = *(const float4*)(w + k);
    float4 x4 = *(const float4*)(xv + k);
    s += wv.x * x4.x + wv.y * x4.y + wv.z * x4.z + wv.w * x4.w;
  }
  h1[b * FFD + o] = 0.5f * s * (1.0f + erff(s * 0.70710678118654752440f));
}

__global__ void k_ffn2(const float* __restrict__ h1, const float* __restrict__ W2,
                       const float* __restrict__ b2, float* __restrict__ x,
                       float* __restrict__ outs, int t, int store_out) {
  int j = blockIdx.x * blockDim.x + threadIdx.x; // < 1024
  int b = j >> 9, h = j & (H - 1);
  const float* w  = W2 + (long)h * FFD;
  const float* hv = h1 + b * FFD;
  float s = b2[h];
  for (int k = 0; k < FFD; k += 4) {
    float4 wv = *(const float4*)(w + k);
    float4 h4 = *(const float4*)(hv + k);
    s += wv.x * h4.x + wv.y * h4.y + wv.z * h4.z + wv.w * h4.w;
  }
  float xn = x[j] + s;
  x[j] = xn;
  if (store_out) outs[((long)b * TT + t) * H + h] = xn;
}

// ---------------- final LN over outs -> bf16 ----------------
__global__ void k_finln(const float* __restrict__ outs, const float* __restrict__ onw,
                        const float* __restrict__ onb, __bf16* __restrict__ Xh) {
  __shared__ float r1[256], r2[256];
  __shared__ float mstat, vstat;
  int row = blockIdx.x, tid = threadIdx.x; // blockDim = 256
  const float* xr = outs + (long)row * H;
  float a0 = xr[tid], a1 = xr[tid + 256];
  r1[tid] = a0 + a1; r2[tid] = a0 * a0 + a1 * a1;
  __syncthreads();
  for (int off = 128; off > 0; off >>= 1) {
    if (tid < off) { r1[tid] += r1[tid + off]; r2[tid] += r2[tid + off]; }
    __syncthreads();
  }
  if (tid == 0) { float m = r1[0] / H; mstat = m; vstat = r2[0] / H - m * m; }
  __syncthreads();
  float rs = rsqrtf(vstat + 1e-5f);
  Xh[(long)row * H + tid]       = (__bf16)((a0 - mstat) * rs * onw[tid] + onb[tid]);
  Xh[(long)row * H + tid + 256] = (__bf16)((a1 - mstat) * rs * onw[tid + 256] + onb[tid + 256]);
}

// ---------------- head GEMM: (64x512) x (32000x512)^T, WMMA bf16 ----------------
__global__ void k_head(const __bf16* __restrict__ Xh, const __bf16* __restrict__ HWb,
                       const float* __restrict__ hb, float* __restrict__ out) {
  int lane = threadIdx.x & 31;
  int wv   = threadIdx.x >> 5;
  int g    = blockIdx.x * 8 + wv;      // 0..1999
  int mtile = g & 3, ngrp = g >> 2;    // 4 x 500 wave tiles of 16x64
  int m0 = mtile * 16, n0 = ngrp * 64;
  int half = lane >> 4;
  const __bf16* arow = Xh + (long)(m0 + (lane & 15)) * H + half * 16;
  const __bf16* brow[4];
  #pragma unroll
  for (int nt = 0; nt < 4; ++nt)
    brow[nt] = HWb + (long)(n0 + nt * 16 + (lane & 15)) * H + half * 16;
  v8f acc[4] = {};
  for (int kk = 0; kk < H; kk += 32) {
    __builtin_prefetch((const void*)(brow[0] + kk + 128), 0, 1);
    v8bf lo = *(const v8bf*)(arow + kk);
    v8bf hi = *(const v8bf*)(arow + kk + 8);
    v16bf a;
    #pragma unroll
    for (int i = 0; i < 8; ++i) { a[i] = lo[i]; a[8 + i] = hi[i]; }
    v16bf vb[4];
    #pragma unroll
    for (int nt = 0; nt < 4; ++nt) vb[nt] = *(const v16bf*)(brow[nt] + kk);
    #pragma unroll
    for (int nt = 0; nt < 4; ++nt)
      acc[nt] = __builtin_amdgcn_wmma_f32_16x16x32_bf16(false, a, false, vb[nt], (short)0,
                                                        acc[nt], false, false);
  }
  int coll = lane & 15;
  #pragma unroll
  for (int nt = 0; nt < 4; ++nt) {
    int col = n0 + nt * 16 + coll;
    #pragma unroll
    for (int r = 0; r < 8; ++r) {
      int m = m0 + r + half * 8;
      out[(long)m * VV + col] = acc[nt][r] + hb[col];
    }
  }
}

extern "C" void kernel_launch(void* const* d_in, const int* in_sizes, int n_in,
                              void* d_out, int out_size, void* d_ws, size_t ws_size,
                              hipStream_t stream) {
  (void)in_sizes; (void)n_in; (void)out_size; (void)ws_size;
  const int*   ids = (const int*)d_in[0];
  const float* st0 = (const float*)d_in[1];
  const float* emb = (const float*)d_in[2];
  const float* nxw = (const float*)d_in[3];
  const float* nxb = (const float*)d_in[4];
  const float* Wp  = (const float*)d_in[5];
  const float* bp  = (const float*)d_in[6];
  const float* Wg  = (const float*)d_in[7];
  const float* Wt  = (const float*)d_in[8];
  const float* nsw = (const float*)d_in[9];
  const float* nsb = (const float*)d_in[10];
  const float* W1  = (const float*)d_in[11];
  const float* b1  = (const float*)d_in[12];
  const float* W2  = (const float*)d_in[13];
  const float* b2  = (const float*)d_in[14];
  const float* onw = (const float*)d_in[15];
  const float* onb = (const float*)d_in[16];
  const float* hW  = (const float*)d_in[17];
  const float* hb  = (const float*)d_in[18];

  char*   ws   = (char*)d_ws;
  float*  stW  = (float*)(ws + OFF_STATES);
  __bf16* hWb  = (__bf16*)(ws + OFF_HEADW);
  __bf16* WgB  = (__bf16*)(ws + OFF_WG);
  __bf16* WtB  = (__bf16*)(ws + OFF_WT);
  float*  x    = (float*)(ws + OFF_X);
  float*  p    = (float*)(ws + OFF_P);
  float*  xln2 = (float*)(ws + OFF_XLN2);
  float*  h1   = (float*)(ws + OFF_H1);
  float*  outs = (float*)(ws + OFF_OUTS);
  __bf16* Xh   = (__bf16*)(ws + OFF_XH);
  float*  gtmp = (float*)(ws + OFF_GTMP);

  // working copy of states (inputs must not be mutated)
  (void)hipMemcpyAsync(stW, st0, sizeof(float) * LL * BB * H * H,
                       hipMemcpyDeviceToDevice, stream);
  // bf16 weight copies (resident in the 192MB L2 thereafter)
  k_cvt_bf16<<<(VV * H / 4 + 255) / 256, 256, 0, stream>>>(hW, hWb, VV * H);
  k_cvt_bf16<<<(LL * H * H / 4 + 255) / 256, 256, 0, stream>>>(Wg, WgB, LL * H * H);
  k_cvt_bf16<<<(LL * H * H / 4 + 255) / 256, 256, 0, stream>>>(Wt, WtB, LL * H * H);

  for (int t = 0; t < TT; ++t) {
    for (int l = 0; l < LL; ++l) {
      float* stl = stW + (long)l * BB * H * H;
      float* g0  = stl + (0 * H + AC) * H;
      float* g1  = stl + (1 * H + AC) * H;
      const __bf16* wgl = WgB + (long)l * H * H;
      const __bf16* wtl = WtB + (long)l * H * H;
      k_pre<<<1, 896, 0, stream>>>(ids, emb, x, nxw + l * H, nxb + l * H,
                                   Wp + (long)l * NP * H, bp + l * NP, p, t, l == 0 ? 1 : 0);
      // recur #1 (ghost -> tmp) fused with both rank-1 active updates
      k_ghost<<<16 + 1536, 256, 0, stream>>>(g0, g1, gtmp, gtmp + 128 * H,
                                             wgl, wtl, stl, p, bp + l * NP, 1);
      // recur #2 (tmp -> ghost)
      k_ghost<<<16, 256, 0, stream>>>(gtmp, gtmp + 128 * H, g0, g1,
                                      wgl, wtl, stl, p, bp + l * NP, 0);
      k_mid<<<BB, 512, 0, stream>>>(stl, x, nsw + l * H, nsb + l * H, xln2);
      k_ffn1<<<16, 256, 0, stream>>>(xln2, W1 + (long)l * FFD * H, b1 + l * FFD, h1);
      k_ffn2<<<4, 256, 0, stream>>>(h1, W2 + (long)l * H * FFD, b2 + l * H, x, outs,
                                    t, l == LL - 1 ? 1 : 0);
    }
  }
  k_finln<<<BB * TT, 256, 0, stream>>>(outs, onw, onb, Xh);
  k_head<<<250, 256, 0, stream>>>(Xh, hWb, hb, (float*)d_out);
  (void)hipMemcpyAsync((float*)d_out + (long)BB * TT * VV, stW,
                       sizeof(float) * LL * BB * H * H, hipMemcpyDeviceToDevice, stream);
}
